// ADDer_40810779246748
// MI455X (gfx1250) — matrix operand order repaired and checked by reading, not compiled, not run on hardware
//
#include <hip/hip_runtime.h>
#include <stdint.h>

typedef __attribute__((ext_vector_type(16))) _Float16 v16h;
typedef __attribute__((ext_vector_type(8)))  _Float16 v8h;
typedef __attribute__((ext_vector_type(4)))  _Float16 v4h;
typedef __attribute__((ext_vector_type(8)))  float    v8f;

#define N_NODES 50000
#define N_EDGES 800000
#define DIM 128
#define MT 5              // m-tiles per wave (3125 = 625 * 5)
#define USE_ASYNC_LDS 1

// ---------------------------------------------------------------- utilities

__global__ void k_zero(float* __restrict__ p, long n) {
  long i = (long)blockIdx.x * blockDim.x + threadIdx.x;
  long s = (long)gridDim.x * blockDim.x;
  for (; i < n; i += s) p[i] = 0.0f;
}

__global__ void k_cvt_f16(const float* __restrict__ src, _Float16* __restrict__ dst, long n) {
  long i = (long)blockIdx.x * blockDim.x + threadIdx.x;
  long s = (long)gridDim.x * blockDim.x;
  for (; i < n; i += s) dst[i] = (_Float16)src[i];
}

// mean = agg / max(cnt,1), emitted in f16
__global__ void k_mean(const float* __restrict__ agg, const float* __restrict__ cnt,
                       _Float16* __restrict__ dst, long n) {
  long i = (long)blockIdx.x * blockDim.x + threadIdx.x;
  long s = (long)gridDim.x * blockDim.x;
  for (; i < n; i += s) {
    long row = i >> 7;  // DIM == 128
    dst[i] = (_Float16)(agg[i] / fmaxf(cnt[row], 1.0f));
  }
}

// -------------------------------------------------- edge scatter (mean aggr)
// One wave per edge; lane l covers features [4l, 4l+4) with f32 atomics.

__global__ void k_scatter_f(const float* __restrict__ x, const int* __restrict__ src,
                            const int* __restrict__ dst, float* __restrict__ agg,
                            float* __restrict__ cnt, int nedges) {
  int wid  = (int)((blockIdx.x * blockDim.x + threadIdx.x) >> 5);
  int lane = threadIdx.x & 31;
  if (wid >= nedges) return;
  int s = src[wid];
  int d = dst[wid];
  float4 v = ((const float4*)(x + (long)s * DIM))[lane];
  float* ap = agg + (long)d * DIM + lane * 4;
  atomicAdd(ap + 0, v.x);
  atomicAdd(ap + 1, v.y);
  atomicAdd(ap + 2, v.z);
  atomicAdd(ap + 3, v.w);
  if (lane == 0) atomicAdd(&cnt[d], 1.0f);
}

__global__ void k_scatter_h(const _Float16* __restrict__ h, const int* __restrict__ src,
                            const int* __restrict__ dst, float* __restrict__ agg,
                            int nedges) {
  int wid  = (int)((blockIdx.x * blockDim.x + threadIdx.x) >> 5);
  int lane = threadIdx.x & 31;
  if (wid >= nedges) return;
  int s = src[wid];
  int d = dst[wid];
  v4h v = *(const v4h*)(h + (long)s * DIM + lane * 4);
  float* ap = agg + (long)d * DIM + lane * 4;
  atomicAdd(ap + 0, (float)v[0]);
  atomicAdd(ap + 1, (float)v[1]);
  atomicAdd(ap + 2, (float)v[2]);
  atomicAdd(ap + 3, (float)v[3]);
}

// -------------------------------------------------------- WMMA fragment I/O

__device__ inline v16h cat8(v8h lo, v8h hi) {
  return __builtin_shufflevector(lo, hi, 0, 1, 2, 3, 4, 5, 6, 7,
                                 8, 9, 10, 11, 12, 13, 14, 15);
}

// A (16x32 f16): lane L holds row M=m0+(L&15); khalf=L>>4 selects K runs
//   elems 0..7  -> K = kk + 8*khalf + j ;  elems 8..15 -> K = kk + 16 + 8*khalf + j
__device__ inline v16h load_a16(const _Float16* __restrict__ Arow, int kk, int khalf) {
  const v8h* p0 = (const v8h*)(Arow + kk + 8 * khalf);
  const v8h* p1 = (const v8h*)(Arow + kk + 16 + 8 * khalf);
  return cat8(*p0, *p1);
}

// Weights are staged into LDS in B-fragment order: chunk c (16 bytes) with
//   c = ((tn*4 + kkb)*32 + lane)*2 + half
// holds W[krow][n0 + half*8 .. +8) where krow = kkb*32 + (lane&15) + 16*(lane>>4).
// Each lane's fragment for (tn,kkb) is then 16 separate 2B LDS reads, one per k row (32B row stride).
__device__ inline void stage_wfrag(_Float16* __restrict__ sdst,
                                   const _Float16* __restrict__ g, int tid) {
#pragma unroll
  for (int i = 0; i < 8; ++i) {
    int c    = i * 256 + tid;        // 2048 chunks of 16B = 32 KB
    int half = c & 1;
    int lane = (c >> 1) & 31;
    int kkb  = (c >> 6) & 3;
    int tn   = c >> 8;
    int krow = kkb * 32 + (lane & 15) + 16 * (lane >> 4);
    const _Float16* src = g + krow * DIM + tn * 16 + half * 8;
    _Float16* dstp = sdst + (long)c * 8;
#if USE_ASYNC_LDS
    unsigned ldsoff = (unsigned)(unsigned long long)dstp;  // flat low-32 == LDS offset
    asm volatile("global_load_async_to_lds_b128 %0, %1, off"
                 :: "v"(ldsoff), "v"(src) : "memory");
#else
    *(v8h*)dstp = *(const v8h*)src;
#endif
  }
}

__device__ inline void stage_fence() {
#if USE_ASYNC_LDS
  asm volatile("s_wait_asynccnt 0" ::: "memory");
#endif
  __syncthreads();
}

__device__ inline v16h load_b_frag(const _Float16* __restrict__ sW, int tn, int kkb,
                                   int lane) {
  const _Float16* base = sW + (long)((tn * 4 + kkb) * 32) * 16 + (lane & 15);
  int kh = 8 * (lane >> 4);
  v16h b;
#pragma unroll
  for (int s = 0; s < 8; ++s) {
    b[s]     = base[(kh + s) * 16];
    b[s + 8] = base[(16 + kh + s) * 16];
  }
  return b;
}

// -------------------------------------------- fused SAGE layer (dual GEMM)
// out = act( meanh @ Wl + selfh @ Wr + bias )
// One wave sweeps MT m-tiles for a fixed n-tile; B fragments live in VGPRs.

__global__ void __launch_bounds__(256)
k_sage_gemm(const _Float16* __restrict__ Amean, const _Float16* __restrict__ Aself,
            const _Float16* __restrict__ Wl, const _Float16* __restrict__ Wr,
            const float* __restrict__ bias,
            _Float16* __restrict__ outh, float* __restrict__ outf, int relu) {
  __shared__ alignas(32) _Float16 sWl[16384];
  __shared__ alignas(32) _Float16 sWr[16384];
  int tid = threadIdx.x;
  stage_wfrag(sWl, Wl, tid);
  stage_wfrag(sWr, Wr, tid);
  stage_fence();

  int wid    = (int)(blockIdx.x * 8 + (tid >> 5));
  int lane   = tid & 31;
  int tn     = wid & 7;              // DIM/16 = 8 n-tiles
  int mgroup = wid >> 3;             // sweeps MT consecutive m-tiles
  int n0     = tn * 16;
  int khalf  = lane >> 4;
  int col    = n0 + (lane & 15);
  float bv   = bias[col];

  // register-cached B fragments: 8 x v16h = 64 VGPRs
  v16h bl[4], br[4];
#pragma unroll
  for (int kkb = 0; kkb < 4; ++kkb) {
    bl[kkb] = load_b_frag(sWl, tn, kkb, lane);
    br[kkb] = load_b_frag(sWr, tn, kkb, lane);
  }

  for (int mt = 0; mt < MT; ++mt) {
    int m0   = (mgroup * MT + mt) * 16;
    int mrow = m0 + (lane & 15);
    const _Float16* Am = Amean + (long)mrow * DIM;
    const _Float16* As = Aself + (long)mrow * DIM;

    v8f acc = {};
#pragma unroll
    for (int kkb = 0; kkb < 4; ++kkb) {
      acc = __builtin_amdgcn_wmma_f32_16x16x32_f16(false, load_a16(Am, kkb * 32, khalf),
              false, bl[kkb], (short)0, acc, false, false);
      acc = __builtin_amdgcn_wmma_f32_16x16x32_f16(false, load_a16(As, kkb * 32, khalf),
              false, br[kkb], (short)0, acc, false, false);
    }

#pragma unroll
    for (int v = 0; v < 8; ++v) {
      int row   = m0 + v + 8 * khalf;
      float val = acc[v] + bv;
      if (relu) val = fmaxf(val, 0.0f);
      if (outh) outh[(long)row * DIM + col] = (_Float16)val;
      if (outf) outf[(long)row * DIM + col] = val;
    }
  }
}

// ------------------------------------------- decoder GEMM + fused BCE terms

__device__ inline float softplusf(float x) {
  if (x > 20.0f)  return x;
  if (x < -20.0f) return __expf(x);
  return log1pf(__expf(x));
}

__global__ void __launch_bounds__(256)
k_dec_loss(const _Float16* __restrict__ embh, const _Float16* __restrict__ Wdec,
           const int* __restrict__ y, float* __restrict__ accum) {
  __shared__ alignas(32) _Float16 sW[16384];
  int tid = threadIdx.x;
  stage_wfrag(sW, Wdec, tid);
  stage_fence();

  int wid    = (int)(blockIdx.x * 8 + (tid >> 5));
  int lane   = tid & 31;
  int tn     = wid & 7;
  int mgroup = wid >> 3;
  int khalf  = lane >> 4;

  v16h bw[4];
#pragma unroll
  for (int kkb = 0; kkb < 4; ++kkb) bw[kkb] = load_b_frag(sW, tn, kkb, lane);

  float pos = 0.0f, neg = 0.0f;
  for (int mt = 0; mt < MT; ++mt) {
    int m0   = (mgroup * MT + mt) * 16;
    int mrow = m0 + (lane & 15);
    const _Float16* Arow = embh + (long)mrow * DIM;

    v8f acc = {};
#pragma unroll
    for (int kkb = 0; kkb < 4; ++kkb) {
      acc = __builtin_amdgcn_wmma_f32_16x16x32_f16(false, load_a16(Arow, kkb * 32, khalf),
              false, bw[kkb], (short)0, acc, false, false);
    }

#pragma unroll
    for (int v = 0; v < 8; ++v) {
      int row   = m0 + v + 8 * khalf;
      float val = acc[v];
      if (y[row] != 0) pos += softplusf(-val);   // bce(logit, 1)
      else             neg += softplusf(val);    // bce(logit, 0)
    }
  }
  atomicAdd(&accum[0], pos);
  atomicAdd(&accum[1], neg);
}

__global__ void k_count_pos(const int* __restrict__ y, float* __restrict__ accum, int n) {
  int i = blockIdx.x * blockDim.x + threadIdx.x;
  int s = gridDim.x * blockDim.x;
  int c = 0;
  for (; i < n; i += s) c += (y[i] != 0) ? 1 : 0;
  if (c) atomicAdd(&accum[2], (float)c);
}

__global__ void k_final(const float* __restrict__ accum, float* __restrict__ loss_out) {
  float npos = accum[2];
  float nneg = (float)N_NODES - npos;
  float l2 = accum[0] / (fmaxf(npos, 1.0f) * (float)DIM);
  float l1 = accum[1] / (fmaxf(nneg, 1.0f) * (float)DIM);
  loss_out[0] = l1 + l2;
}

// ---------------------------------------------------------------- launcher

extern "C" void kernel_launch(void* const* d_in, const int* in_sizes, int n_in,
                              void* d_out, int out_size, void* d_ws, size_t ws_size,
                              hipStream_t stream) {
  (void)in_sizes; (void)n_in; (void)out_size; (void)ws_size;
  const float* x    = (const float*)d_in[0];
  const int*  edge  = (const int*)d_in[1];   // [2,E]: row0 = src, row1 = dst
  const int*  y     = (const int*)d_in[2];
  // d_in[3] = pretrain (unused in eval path)
  const float* W1l  = (const float*)d_in[4];
  const float* b1l  = (const float*)d_in[5];
  const float* W1r  = (const float*)d_in[6];
  const float* W2l  = (const float*)d_in[7];
  const float* b2l  = (const float*)d_in[8];
  const float* W2r  = (const float*)d_in[9];
  const float* Wdec = (const float*)d_in[10];

  float* agg  = (float*)d_ws;                      // N*DIM f32
  float* cnt  = agg + (size_t)N_NODES * DIM;       // N f32
  float* scal = cnt + N_NODES;                     // 4 f32 (pos, neg, npos)
  _Float16* xh   = (_Float16*)(scal + 4);          // N*DIM f16 each:
  _Float16* h1h  = xh   + (size_t)N_NODES * DIM;
  _Float16* mh   = h1h  + (size_t)N_NODES * DIM;
  _Float16* embh = mh   + (size_t)N_NODES * DIM;
  _Float16* w1lh = embh + (size_t)N_NODES * DIM;   // 5 x DIM*DIM f16
  _Float16* w1rh = w1lh + DIM * DIM;
  _Float16* w2lh = w1rh + DIM * DIM;
  _Float16* w2rh = w2lh + DIM * DIM;
  _Float16* wdh  = w2rh + DIM * DIM;

  const int* src = edge;
  const int* dst = edge + N_EDGES;
  float* out = (float*)d_out;

  const int eblocks    = (N_EDGES * 32 + 255) / 256;              // one wave / edge
  const int gemmblocks = (N_NODES / 16 / MT) * (DIM / 16) / 8;    // 625

  // ---- f16 operand precompute (once per launch)
  k_cvt_f16<<<4096, 256, 0, stream>>>(x, xh, (long)N_NODES * DIM);
  k_cvt_f16<<<64, 256, 0, stream>>>(W1l, w1lh, DIM * DIM);
  k_cvt_f16<<<64, 256, 0, stream>>>(W1r, w1rh, DIM * DIM);
  k_cvt_f16<<<64, 256, 0, stream>>>(W2l, w2lh, DIM * DIM);
  k_cvt_f16<<<64, 256, 0, stream>>>(W2r, w2rh, DIM * DIM);
  k_cvt_f16<<<64, 256, 0, stream>>>(Wdec, wdh, DIM * DIM);

  // ---- layer 1
  k_zero<<<2048, 256, 0, stream>>>(agg, (long)N_NODES * DIM + N_NODES + 4);
  k_scatter_f<<<eblocks, 256, 0, stream>>>(x, src, dst, agg, cnt, N_EDGES);
  k_mean<<<4096, 256, 0, stream>>>(agg, cnt, mh, (long)N_NODES * DIM);
  k_sage_gemm<<<gemmblocks, 256, 0, stream>>>(mh, xh, w1lh, w1rh, b1l,
                                              h1h, nullptr, 1);

  // ---- layer 2 (cnt depends only on edge structure: reuse)
  k_zero<<<2048, 256, 0, stream>>>(agg, (long)N_NODES * DIM);
  k_scatter_h<<<eblocks, 256, 0, stream>>>(h1h, src, dst, agg, N_EDGES);
  k_mean<<<4096, 256, 0, stream>>>(agg, cnt, mh, (long)N_NODES * DIM);
  k_sage_gemm<<<gemmblocks, 256, 0, stream>>>(mh, h1h, w2lh, w2rh, b2l,
                                              embh, out + 1, 0);

  // ---- decoder + loss
  k_count_pos<<<256, 256, 0, stream>>>(y, scal, N_NODES);
  k_dec_loss<<<gemmblocks, 256, 0, stream>>>(embh, wdh, y, scal);
  k_final<<<1, 1, 0, stream>>>(scal, out);
}
